// Decoder_11089605558949
// MI455X (gfx1250) — matrix operand order repaired and checked by reading, not compiled, not run on hardware
//
#include <hip/hip_runtime.h>
#include <hip/hip_bf16.h>
#include <cstdint>

// ---------------------------------------------------------------------------
// Types for CDNA5 WMMA (wave32, 16x16x32 bf16 -> f32)
// ---------------------------------------------------------------------------
typedef __attribute__((ext_vector_type(16))) __bf16 v16bf;
typedef __attribute__((ext_vector_type(8)))  __bf16 bf16x8;
typedef __attribute__((ext_vector_type(8)))  float  v8f;

namespace {
constexpr int B = 64, T = 8, S = 1024, V = 32000, E = 256, U = 512;
constexpr int BT = B * T;   // 512
constexpr int BS = B * S;   // 65536
}

__device__ __forceinline__ float fast_exp(float x) {
    return __builtin_amdgcn_exp2f(x * 1.4426950408889634f);
}
__device__ __forceinline__ float fast_tanh(float x) {
    float xc = fminf(fmaxf(x, -15.f), 15.f);
    float e  = __builtin_amdgcn_exp2f(xc * 2.8853900817779268f); // e^(2x)
    return (e - 1.f) / (e + 1.f);
}
__device__ __forceinline__ float fast_sigmoid(float x) {
    return 1.f / (1.f + __builtin_amdgcn_exp2f(-x * 1.4426950408889634f));
}
__device__ __forceinline__ __bf16 f32_to_bf16(float f) {
    unsigned u = __float_as_uint(f);
    u += 0x7FFFu + ((u >> 16) & 1u);            // round-to-nearest-even
    unsigned short h = (unsigned short)(u >> 16);
    union { unsigned short u16; __bf16 b; } cv; cv.u16 = h;
    return cv.b;
}

// ---------------------------------------------------------------------------
// Small utility kernels
// ---------------------------------------------------------------------------
__global__ void k_convert_bf16(const float* __restrict__ src,
                               __bf16* __restrict__ dst, int n) {
    int i = blockIdx.x * blockDim.x + threadIdx.x;
    if (i < n) dst[i] = f32_to_bf16(src[i]);
}

__global__ void k_zero_f32(float* __restrict__ p, int n) {
    int i = blockIdx.x * blockDim.x + threadIdx.x;
    if (i < n) p[i] = 0.f;
}

// dst[C][R] (bf16) = transpose of src[R][C] (f32); R,C multiples of 32
__global__ void k_transpose_bf16(const float* __restrict__ src,
                                 __bf16* __restrict__ dst, int R, int C) {
    __shared__ float tile[32][33];
    int c0 = blockIdx.x * 32, r0 = blockIdx.y * 32;
    int tx = threadIdx.x, ty = threadIdx.y; // 32 x 8
#pragma unroll
    for (int i = 0; i < 32; i += 8)
        tile[ty + i][tx] = src[(size_t)(r0 + ty + i) * C + (c0 + tx)];
    __syncthreads();
#pragma unroll
    for (int i = 0; i < 32; i += 8)
        dst[(size_t)(c0 + ty + i) * R + (r0 + tx)] = f32_to_bf16(tile[tx][ty + i]);
}

// vectors = emb[new_tokens] -> bf16 [BT][E]
__global__ void k_embed(const int* __restrict__ tok,
                        const float* __restrict__ emb,
                        __bf16* __restrict__ dst) {
    int row = blockIdx.x;      // b*T + t  in [0, BT)
    int e   = threadIdx.x;     // [0, E)
    int id  = tok[row];
    dst[(size_t)row * E + e] = f32_to_bf16(emb[(size_t)id * E + e]);
}

// ---------------------------------------------------------------------------
// bf16 WMMA GEMM:  C[M][N] = act( A[M][K] @ B + bias ),  Bt is B^T [N][K]
// Each wave owns a 16x128 C slab (8 WMMA tiles reusing one A fragment).
// K is a template parameter -> fully unrolled k-loop with immediate load
// offsets, all 8 B fragments loaded per k-step before the WMMA chain (one
// load clause + one wait instead of a wait per WMMA).
// M%16==0, N%128==0. act: 0=none, 1=tanh. C16 optional bf16 copy of C.
// ---------------------------------------------------------------------------
template <int K>
__global__ void k_gemm_bf16_wmma(const __bf16* __restrict__ A,
                                 const __bf16* __restrict__ Bt,
                                 const float* __restrict__ bias,
                                 float* __restrict__ C,
                                 __bf16* __restrict__ C16,
                                 int M, int N, int act) {
    constexpr int NT = 8;
    int waveId  = (int)((blockIdx.x * blockDim.x + threadIdx.x) >> 5);
    int mTiles  = M >> 4;
    int nChunks = N / (16 * NT);
    if (waveId >= mTiles * nChunks) return;      // wave-uniform: EXEC stays all-1

    int mTile  = waveId % mTiles;
    int nChunk = waveId / mTiles;
    int lane   = threadIdx.x & 31;
    int r      = lane & 15;
    int grp    = lane >> 4;

    // A fragment rows: lanes 0-15 hold K..K+15, lanes 16-31 hold K+16..K+31
    const __bf16* Arow = A + (size_t)(mTile * 16 + r) * K + grp * 16;
    // B fragment rows: lanes 0-15 hold K..K+15, lanes 16-31 hold K+16..K+31
    const __bf16* Brow[NT];
#pragma unroll
    for (int n = 0; n < NT; ++n) {
        int col = (nChunk * NT + n) * 16 + r;
        Brow[n] = Bt + (size_t)col * K + grp * 16;
    }

    v8f acc[NT] = {};

#pragma unroll
    for (int k0 = 0; k0 < K; k0 += 32) {
        bf16x8 a0 = *(const bf16x8*)(Arow + k0);
        bf16x8 a1 = *(const bf16x8*)(Arow + k0 + 8);
        v16bf bfr[NT];
#pragma unroll
        for (int n = 0; n < NT; ++n)
            bfr[n] = *(const v16bf*)(Brow[n] + k0);

        v16bf a;
#pragma unroll
        for (int i = 0; i < 8; ++i) { a[i] = a0[i]; a[8 + i] = a1[i]; }

#pragma unroll
        for (int n = 0; n < NT; ++n)
            acc[n] = __builtin_amdgcn_wmma_f32_16x16x32_bf16(
                false, a, false, bfr[n], (short)0, acc[n], false, false);
    }

#pragma unroll
    for (int n = 0; n < NT; ++n) {
        int col = (nChunk * NT + n) * 16 + r;
        float bv = bias ? bias[col] : 0.f;
#pragma unroll
        for (int e = 0; e < 8; ++e) {
            int row = mTile * 16 + grp * 8 + e;
            float v = acc[n][e] + bv;
            if (act == 1) v = fast_tanh(v);
            size_t idx = (size_t)row * N + col;
            C[idx] = v;
            if (C16) C16[idx] = f32_to_bf16(v);
        }
    }
}

// ---------------------------------------------------------------------------
// GRU step (Keras reset_after=True, gates [z,r,h]); one block per batch elem.
// rec = h @ RK + b_r ; z=sig(xz+rz); r=sig(xr+rr); hh=tanh(xh+r*rh)
// ---------------------------------------------------------------------------
__global__ void k_gru_step(const float* __restrict__ xproj,   // [BT][3U]
                           const float* __restrict__ RK,      // [U][3U]
                           const float* __restrict__ gbias,   // [2][3U]
                           float* __restrict__ h,             // [B][U]
                           float* __restrict__ rnnout,        // [BT][U]
                           int t) {
    __shared__ float hs[U];
    int b = blockIdx.x, j = threadIdx.x;       // 512 threads
    hs[j] = h[(size_t)b * U + j];
    __syncthreads();

    const float* br = gbias + 3 * U;
    float rz = br[j], rr = br[U + j], rh = br[2 * U + j];
    for (int k = 0; k < U; ++k) {
        float hk = hs[k];
        const float* w = RK + (size_t)k * (3 * U);
        rz = fmaf(hk, w[j],         rz);
        rr = fmaf(hk, w[U + j],     rr);
        rh = fmaf(hk, w[2 * U + j], rh);
    }
    const float* xp = xproj + (size_t)(b * T + t) * (3 * U);
    float z  = fast_sigmoid(xp[j] + rz);
    float r  = fast_sigmoid(xp[U + j] + rr);
    float hh = fast_tanh(xp[2 * U + j] + r * rh);
    float hn = z * hs[j] + (1.f - z) * hh;
    h[(size_t)b * U + j] = hn;
    rnnout[(size_t)(b * T + t) * U + j] = hn;
}

// ---------------------------------------------------------------------------
// Additive-attention scores: scores[b,t,s] = sum_u scale_u*tanh(q[b,t,u]+k[b,s,u])
// Block = (b, 256-wide s chunk); k row read once, reused for all 8 t.
// ---------------------------------------------------------------------------
__global__ void k_scores(const float* __restrict__ q,      // [BT][U]
                         const float* __restrict__ kmat,   // [BS][U]
                         const float* __restrict__ scale,  // [U]
                         const unsigned char* __restrict__ mask, // [B][S]
                         float* __restrict__ out) {        // [BT][S]
    __shared__ float qs[T][U];
    __shared__ float ss[U];
    constexpr int CH = S / 256;                // 4 chunks per batch
    int b = blockIdx.x / CH, chunk = blockIdx.x % CH;
    int tid = threadIdx.x;

    for (int i = tid; i < T * U; i += 256) qs[i / U][i % U] = q[(size_t)b * T * U + i];
    for (int i = tid; i < U;     i += 256) ss[i] = scale[i];
    __syncthreads();

    int s = chunk * 256 + tid;
    const float* krow = kmat + ((size_t)b * S + s) * U;
    float acc[T];
#pragma unroll
    for (int tt = 0; tt < T; ++tt) acc[tt] = 0.f;

    for (int u = 0; u < U; u += 4) {
        float4 kv4 = *(const float4*)(krow + u);
        float kv[4] = {kv4.x, kv4.y, kv4.z, kv4.w};
#pragma unroll
        for (int uu = 0; uu < 4; ++uu) {
            float sc = ss[u + uu];
            float kk = kv[uu];
#pragma unroll
            for (int tt = 0; tt < T; ++tt)
                acc[tt] = fmaf(sc, fast_tanh(qs[tt][u + uu] + kk), acc[tt]);
        }
    }
    bool ok = mask[(size_t)b * S + s] != 0;
#pragma unroll
    for (int tt = 0; tt < T; ++tt)
        out[((size_t)(b * T + tt)) * S + s] = ok ? acc[tt] : -1e9f;
}

// In-place softmax over rows of length S; one block (256 thr) per row.
__global__ void k_softmax(float* __restrict__ data) {
    __shared__ float red[256];
    float* p = data + (size_t)blockIdx.x * S;
    int tid = threadIdx.x;
    float v[4], m = -1e30f;
#pragma unroll
    for (int i = 0; i < 4; ++i) { v[i] = p[tid + i * 256]; m = fmaxf(m, v[i]); }
    red[tid] = m; __syncthreads();
    for (int o = 128; o > 0; o >>= 1) {
        if (tid < o) red[tid] = fmaxf(red[tid], red[tid + o]);
        __syncthreads();
    }
    m = red[0]; __syncthreads();
    float sum = 0.f;
#pragma unroll
    for (int i = 0; i < 4; ++i) { v[i] = fast_exp(v[i] - m); sum += v[i]; }
    red[tid] = sum; __syncthreads();
    for (int o = 128; o > 0; o >>= 1) {
        if (tid < o) red[tid] += red[tid + o];
        __syncthreads();
    }
    float inv = 1.f / red[0];
#pragma unroll
    for (int i = 0; i < 4; ++i) p[tid + i * 256] = v[i] * inv;
}

// context[b,t,:] = attn_w[b,t,:] @ enc[b]; writes concat=[context|rnn_out] bf16
__global__ void k_context_concat(const float* __restrict__ attn,   // [BT][S]
                                 const float* __restrict__ enc,    // [B][S][U]
                                 const float* __restrict__ rnnout, // [BT][U]
                                 __bf16* __restrict__ concat16) {  // [BT][2U]
    __shared__ float aw[T][S];                 // 32 KB
    int b = blockIdx.x, tid = threadIdx.x;     // 512 threads
    for (int i = tid; i < T * S; i += U) aw[i / S][i % S] = attn[(size_t)b * T * S + i];
    __syncthreads();

    int u = tid;
    float acc[T];
#pragma unroll
    for (int tt = 0; tt < T; ++tt) acc[tt] = 0.f;
    const float* encb = enc + (size_t)b * S * U;
    for (int s = 0; s < S; ++s) {
        float e = encb[(size_t)s * U + u];
#pragma unroll
        for (int tt = 0; tt < T; ++tt) acc[tt] = fmaf(aw[tt][s], e, acc[tt]);
    }
#pragma unroll
    for (int tt = 0; tt < T; ++tt) {
        size_t row = (size_t)b * T + tt;
        concat16[row * (2 * U) + u]     = f32_to_bf16(acc[tt]);
        concat16[row * (2 * U) + U + u] = f32_to_bf16(rnnout[row * U + u]);
    }
}

// ---------------------------------------------------------------------------
// Host: orchestrate the whole decoder step on `stream` (graph-capture safe).
// ---------------------------------------------------------------------------
extern "C" void kernel_launch(void* const* d_in, const int* in_sizes, int n_in,
                              void* d_out, int out_size, void* d_ws, size_t ws_size,
                              hipStream_t stream) {
    (void)in_sizes; (void)n_in; (void)out_size; (void)ws_size;

    const int*   tok   = (const int*)d_in[0];
    const float* enc   = (const float*)d_in[1];
    const unsigned char* mask = (const unsigned char*)d_in[2];
    const float* emb   = (const float*)d_in[3];
    const float* gruK  = (const float*)d_in[4];
    const float* gruRK = (const float*)d_in[5];
    const float* gruB  = (const float*)d_in[6];
    const float* W1    = (const float*)d_in[7];
    const float* W2    = (const float*)d_in[8];
    const float* scale = (const float*)d_in[9];
    const float* Wc    = (const float*)d_in[10];
    const float* fcW   = (const float*)d_in[11];
    const float* fcb   = (const float*)d_in[12];

    float* out_logits = (float*)d_out;                      // [BT][V]
    float* out_attn   = out_logits + (size_t)BT * V;        // [BT][S]
    float* out_state  = out_attn + (size_t)BT * S;          // [B][U]

    char* ws = (char*)d_ws;
    size_t off = 0;
    auto take = [&](size_t bytes) -> char* {
        char* p = ws + off;
        off += (bytes + 255) & ~(size_t)255;
        return p;
    };
    __bf16* enc16    = (__bf16*)take((size_t)BS * U * 2);       // 64 MB
    float*  kmat     = (float*) take((size_t)BS * U * 4);       // 128 MB
    __bf16* fcWt     = (__bf16*)take((size_t)V * U * 2);        // 31 MB
    __bf16* W2t      = (__bf16*)take((size_t)U * U * 2);
    __bf16* W1t      = (__bf16*)take((size_t)U * U * 2);
    __bf16* gruKt    = (__bf16*)take((size_t)(3 * U) * E * 2);
    __bf16* Wct      = (__bf16*)take((size_t)U * (2 * U) * 2);
    __bf16* vec16    = (__bf16*)take((size_t)BT * E * 2);
    float*  xproj    = (float*) take((size_t)BT * 3 * U * 4);
    float*  hbuf     = (float*) take((size_t)B * U * 4);
    float*  rnnout   = (float*) take((size_t)BT * U * 4);
    __bf16* rnn16    = (__bf16*)take((size_t)BT * U * 2);
    float*  qbuf     = (float*) take((size_t)BT * U * 4);
    __bf16* concat16 = (__bf16*)take((size_t)BT * 2 * U * 2);
    float*  av32     = (float*) take((size_t)BT * U * 4);
    __bf16* av16     = (__bf16*)take((size_t)BT * U * 2);

    auto gemm = [&](const __bf16* A, const __bf16* Bt, const float* bias,
                    float* C, __bf16* C16, int M, int N, int K, int act) {
        int waves  = (M / 16) * (N / 128);
        int blocks = (waves + 7) / 8;           // 8 waves (256 threads) / block
        switch (K) {
        case 256:
            k_gemm_bf16_wmma<256><<<blocks, 256, 0, stream>>>(A, Bt, bias, C, C16, M, N, act);
            break;
        case 512:
            k_gemm_bf16_wmma<512><<<blocks, 256, 0, stream>>>(A, Bt, bias, C, C16, M, N, act);
            break;
        case 1024:
            k_gemm_bf16_wmma<1024><<<blocks, 256, 0, stream>>>(A, Bt, bias, C, C16, M, N, act);
            break;
        }
    };

    // 1) Precision conversion / weight transposition (bf16, B stored as [N][K])
    { int n = BS * U; k_convert_bf16<<<(n + 255) / 256, 256, 0, stream>>>(enc, enc16, n); }
    dim3 tb(32, 8);
    k_transpose_bf16<<<dim3(U / 32, U / 32),       tb, 0, stream>>>(W2,  W2t,  U, U);
    k_transpose_bf16<<<dim3(U / 32, U / 32),       tb, 0, stream>>>(W1,  W1t,  U, U);
    k_transpose_bf16<<<dim3((3 * U) / 32, E / 32), tb, 0, stream>>>(gruK, gruKt, E, 3 * U);
    k_transpose_bf16<<<dim3(U / 32, (2 * U) / 32), tb, 0, stream>>>(Wc,  Wct,  2 * U, U);
    k_transpose_bf16<<<dim3(V / 32, U / 32),       tb, 0, stream>>>(fcW, fcWt, U, V);

    // 2) Embedding gather + GRU input projection (WMMA) + recurrence (8 steps)
    k_embed<<<BT, E, 0, stream>>>(tok, emb, vec16);
    gemm(vec16, gruKt, gruB /* b_i */, xproj, nullptr, BT, 3 * U, E, 0);
    k_zero_f32<<<(B * U + 255) / 256, 256, 0, stream>>>(hbuf, B * U);
    for (int t = 0; t < T; ++t)
        k_gru_step<<<B, U, 0, stream>>>(xproj, gruRK, gruB, hbuf, rnnout, t);
    { int n = BT * U; k_convert_bf16<<<(n + 255) / 256, 256, 0, stream>>>(rnnout, rnn16, n); }

    // 3) Attention projections (WMMA): q small, k is the 34-GFLOP GEMM
    gemm(rnn16, W1t, nullptr, qbuf, nullptr, BT, U, U, 0);
    gemm(enc16, W2t, nullptr, kmat, nullptr, BS, U, U, 0);

    // 4) Additive scores (268M tanh) + masked softmax, in-place in d_out
    k_scores<<<B * (S / 256), 256, 0, stream>>>(qbuf, kmat, scale, mask, out_attn);
    k_softmax<<<BT, 256, 0, stream>>>(out_attn);

    // 5) Context einsum + concat(bf16), then Wc (tanh) and vocab GEMMs (WMMA)
    k_context_concat<<<B, U, 0, stream>>>(out_attn, enc, rnnout, concat16);
    gemm(concat16, Wct, nullptr, av32, av16, BT, U, 2 * U, 1 /* tanh */);
    gemm(av16, fcWt, fcb, out_logits, nullptr, BT, V, U, 0);

    // 6) Final GRU state
    hipMemcpyAsync(out_state, hbuf, (size_t)B * U * sizeof(float),
                   hipMemcpyDeviceToDevice, stream);
}
